// PseudoTokenGridEncoder_25194278159165
// MI455X (gfx1250) — matrix operand order repaired and checked by reading, not compiled, not run on hardware
//
#include <hip/hip_runtime.h>
#include <hip/hip_bf16.h>
#include <stddef.h>

// Problem constants (match reference)
#define B_    8
#define U_    8192
#define N0_   64
#define N1_   64
#define S_    4096          // N0*N1
#define E_    128
#define H_    8
#define DH_   16
#define MAXP  64            // patch capacity (expected max ~12 for 8192 balls / 4096 bins)
#define NCELL (B_*S_)       // 32768
#define NTOK  (B_*U_)       // 65536

typedef __attribute__((ext_vector_type(16))) _Float16 v16h;
typedef __attribute__((ext_vector_type(8)))  float    v8f;

#define WFRAG (4 * 8 * 32 * 16)   // f16 elements for one staged 128x128 W

// f32 -> f16: single v_cvt_f16_f32 (RTE), vs ~3 VALU ops for software bf16.
static __device__ __forceinline__ _Float16 f2h(float f) { return (_Float16)f; }

// Stage W (128x128 row-major f32) into B-fragment layout in LDS:
// fragment (kc,nt): lane<16, elem e <-> W[kc*32 + (e<8 ? e : 16+e-8)][nt*16 + lane]
//                   lane>=16, elem e <-> W[kc*32 + 8 + (e<8 ? e : 16+e-8)][nt*16 + lane-16]
static __device__ __forceinline__ void stage_w(const float* __restrict__ W,
                                               _Float16* Wl, int tid) {
  for (int i4 = tid; i4 < (E_ * E_) / 4; i4 += 128) {
    int i = i4 << 2;
    int k = i >> 7, n = i & 127;          // n multiple of 4
    int kc = k >> 5, ko = k & 31;
    int nt = n >> 4;
    int e = (ko & 7) | ((ko >> 4) << 3);
    float4 w4 = *(const float4*)&W[i];
    float f[4] = {w4.x, w4.y, w4.z, w4.w};
#pragma unroll
    for (int j = 0; j < 4; ++j) {
      int l = ((n + j) & 15) | (((ko >> 3) & 1) << 4);
      Wl[((((kc << 3) | nt) << 5 | l) << 4) | e] = f2h(f[j]);
    }
  }
}

// Build this lane's 4 A fragments (one per kc) directly in registers.
// A-fragment: lane<16 row r=lane: e=0..7 <-> K=kc*32+0..7, e=8..15 <-> K=kc*32+16..23
//             lane>=16 row r=lane-16: K offsets +8 (i.e. 8..15 and 24..31)
static __device__ __forceinline__ void load_a_frags(
    const float* __restrict__ A, long base, int lane, v16h af[4]) {
  int r = lane & 15;
  int koff = (lane >> 4) << 3;   // 0 or 8
  const float* row = A + (base + r) * E_;
#pragma unroll
  for (int kc = 0; kc < 4; ++kc) {
    const float* p0 = row + kc * 32 + koff;
    float4 q0 = *(const float4*)(p0);
    float4 q1 = *(const float4*)(p0 + 4);
    float4 q2 = *(const float4*)(p0 + 16);
    float4 q3 = *(const float4*)(p0 + 20);
    v16h a;
    a[0] = f2h(q0.x);  a[1] = f2h(q0.y);  a[2] = f2h(q0.z);  a[3] = f2h(q0.w);
    a[4] = f2h(q1.x);  a[5] = f2h(q1.y);  a[6] = f2h(q1.z);  a[7] = f2h(q1.w);
    a[8] = f2h(q2.x);  a[9] = f2h(q2.y);  a[10] = f2h(q2.z); a[11] = f2h(q2.w);
    a[12] = f2h(q3.x); a[13] = f2h(q3.y); a[14] = f2h(q3.z); a[15] = f2h(q3.w);
    af[kc] = a;
  }
}

// 16x128 strip: accumulate over K=128 (4 kc chunks), 8 N-tiles, then store C.
static __device__ __forceinline__ void gemm_strip(
    const _Float16* Wl, const v16h af[4], float* __restrict__ C,
    long base, int lane) {
  v8f c[8];
#pragma unroll
  for (int nt = 0; nt < 8; ++nt)
#pragma unroll
    for (int v = 0; v < 8; ++v) c[nt][v] = 0.0f;

#pragma unroll
  for (int kc = 0; kc < 4; ++kc) {
#pragma unroll
    for (int nt = 0; nt < 8; ++nt) {
      v16h b = *(const v16h*)&Wl[((((kc << 3) | nt) << 5) | lane) << 4];
      c[nt] = __builtin_amdgcn_wmma_f32_16x16x32_f16(
          false, af[kc], false, b, (short)0, c[nt], false, false);
    }
  }
  // C fragment: lane<16 -> rows M=0..7 (reg v), N=lane; lane>=16 -> M=8..15
  const int n0 = lane & 15;
  const int mo = (lane >> 4) * 8;
#pragma unroll
  for (int nt = 0; nt < 8; ++nt)
#pragma unroll
    for (int v = 0; v < 8; ++v)
      C[(base + mo + v) * E_ + nt * 16 + n0] = c[nt][v];
}

// ---------------------------------------------------------------------------
// C(M x 128) = A(M x 128) @ W(128 x 128). Grid-stride over 64-row strips;
// W staged to LDS once per block (compiler keeps fragments register-resident).
// 4 waves/block, 16 rows/wave/strip. Uses v_wmma_f32_16x16x32_f16.
// ---------------------------------------------------------------------------
__global__ __launch_bounds__(128) void gemm_f16_128(
    const float* __restrict__ A, const float* __restrict__ W,
    float* __restrict__ C, int nStrips) {
  __shared__ alignas(32) _Float16 Wl[WFRAG];   // 32 KB
  const int tid = threadIdx.x, wave = tid >> 5, lane = tid & 31;
  stage_w(W, Wl, tid);
  __syncthreads();
  for (int strip = blockIdx.x; strip < nStrips; strip += gridDim.x) {
    long base = (long)strip * 64 + wave * 16;
    v16h af[4];
    load_a_frags(A, base, lane, af);
    gemm_strip(Wl, af, C, base, lane);
  }
}

// Dual-output variant: C0 = A@W0, C1 = A@W1 — A read from HBM once,
// register-resident A fragments reused for both weight matrices.
__global__ __launch_bounds__(128) void gemm_f16_128_dual(
    const float* __restrict__ A, const float* __restrict__ W0,
    const float* __restrict__ W1, float* __restrict__ C0,
    float* __restrict__ C1, int nStrips) {
  __shared__ alignas(32) _Float16 Wl[2 * WFRAG];   // 64 KB (of 320 KB/WGP)
  const int tid = threadIdx.x, wave = tid >> 5, lane = tid & 31;
  stage_w(W0, Wl, tid);
  stage_w(W1, Wl + WFRAG, tid);
  __syncthreads();
  for (int strip = blockIdx.x; strip < nStrips; strip += gridDim.x) {
    long base = (long)strip * 64 + wave * 16;
    v16h af[4];
    load_a_frags(A, base, lane, af);
    gemm_strip(Wl, af, C0, base, lane);
    gemm_strip(Wl + WFRAG, af, C1, base, lane);
  }
}

// ---------------------------------------------------------------------------
// Helper kernels
// ---------------------------------------------------------------------------
__global__ void k_zero(int* __restrict__ p, int n) {
  int i = blockIdx.x * blockDim.x + threadIdx.x;
  if (i < n) p[i] = 0;
}

__global__ void k_copy4(const float4* __restrict__ src, float4* __restrict__ dst, int n4) {
  int i = blockIdx.x * blockDim.x + threadIdx.x;
  if (i < n4) dst[i] = src[i];
}

// Bin tokens to nearest grid cell; assign a slot via atomicAdd.
// Order within a cell differs from reference cumcount, but softmax attention
// over the kv set is permutation invariant, so the result is identical.
__global__ void k_bin(const float* __restrict__ xc,
                      int* __restrict__ counts, int* __restrict__ ctok,
                      int* __restrict__ cellid) {
  int t = blockIdx.x * blockDim.x + threadIdx.x;
  if (t >= NTOK) return;
  int b = t / U_;
  float x0 = xc[t * 2 + 0], x1 = xc[t * 2 + 1];
  // grid = linspace(-1,1,64); spacing 2/63; idx = round((x+1)*31.5)
  int i0 = (int)rintf((x0 + 1.0f) * 31.5f);
  int i1 = (int)rintf((x1 + 1.0f) * 31.5f);
  i0 = min(max(i0, 0), N0_ - 1);
  i1 = min(max(i1, 0), N1_ - 1);
  int cell = b * S_ + i0 * N1_ + i1;
  cellid[t] = cell;
  int slot = atomicAdd(&counts[cell], 1);
  if (slot < MAXP) ctok[(size_t)cell * MAXP + slot] = t;
}

// kfake = fake @ Wk, vfake = fake @ Wv  (one 128-thread block)
__global__ void k_fake(const float* __restrict__ fake,
                       const float* __restrict__ Wk, const float* __restrict__ Wv,
                       float* __restrict__ kf, float* __restrict__ vf) {
  int n = threadIdx.x;
  float ak = 0.f, av = 0.f;
  for (int k = 0; k < E_; ++k) {
    float f = fake[k];
    ak += f * Wk[k * E_ + n];
    av += f * Wv[k * E_ + n];
  }
  kf[n] = ak; vf[n] = av;
}

// flog[s,h] = dot(QH[s, h*16:], kfake[h*16:]) / sqrt(16) — batch-shared
__global__ void k_flogit(const float* __restrict__ QH,
                         const float* __restrict__ kfake,
                         float* __restrict__ flog) {
  int g = blockIdx.x * blockDim.x + threadIdx.x;
  if (g >= S_ * H_) return;
  int s = g >> 3, h = g & 7;
  const float* q = QH + (size_t)s * E_ + h * DH_;
  const float* k = kfake + h * DH_;
  float acc = 0.f;
#pragma unroll
  for (int j = 0; j < DH_; ++j) acc += q[j] * k[j];
  flog[g] = acc * 0.25f;   // 1/sqrt(dh), dh=16
}

// logits[t,h] = dot(QH[s(t), h*16:], K[t, h*16:]) / 4
__global__ void k_logits(const float* __restrict__ QH, const float* __restrict__ K,
                         const int* __restrict__ cellid, float* __restrict__ logits) {
  int g = blockIdx.x * blockDim.x + threadIdx.x;
  if (g >= NTOK * H_) return;
  int t = g >> 3, h = g & 7;
  int s = cellid[t] & (S_ - 1);
  const float* q = QH + (size_t)s * E_ + h * DH_;
  const float* k = K + (size_t)t * E_ + h * DH_;
  float acc = 0.f;
#pragma unroll
  for (int j = 0; j < DH_; ++j) acc += q[j] * k[j];
  logits[g] = acc * 0.25f;
}

// One wave per cell: per-head softmax over (cnt tokens + fake), weighted V sum.
// lane handles 4 consecutive dims (d0 = lane*4), head = lane>>2.
__global__ __launch_bounds__(256) void k_attn(
    const int* __restrict__ counts, const int* __restrict__ ctok,
    const float* __restrict__ logits, const float* __restrict__ flog,
    const float* __restrict__ V, const float* __restrict__ vfake,
    float* __restrict__ OutH) {
  int wave = threadIdx.x >> 5;
  int lane = threadIdx.x & 31;
  int cell = blockIdx.x * 8 + wave;
  int s = cell & (S_ - 1);
  int h = lane >> 2;
  int d0 = lane << 2;
  int cnt = counts[cell];
  if (cnt > MAXP) cnt = MAXP;
  const int* list = ctok + (size_t)cell * MAXP;

  float fl = flog[s * H_ + h];
  float m = fl;
  for (int i = 0; i < cnt; ++i)
    m = fmaxf(m, logits[list[i] * H_ + h]);

  float wf = __expf(fl - m);
  float denom = wf;
  float a0 = wf * vfake[d0 + 0];
  float a1 = wf * vfake[d0 + 1];
  float a2 = wf * vfake[d0 + 2];
  float a3 = wf * vfake[d0 + 3];
  for (int i = 0; i < cnt; ++i) {
    int t = list[i];
    float w = __expf(logits[t * H_ + h] - m);
    denom += w;
    const float* v = V + (size_t)t * E_ + d0;
    a0 += w * v[0]; a1 += w * v[1]; a2 += w * v[2]; a3 += w * v[3];
  }
  float inv = 1.0f / denom;
  float* o = OutH + (size_t)cell * E_ + d0;
  o[0] = a0 * inv; o[1] = a1 * inv; o[2] = a2 * inv; o[3] = a3 * inv;
}

// ---------------------------------------------------------------------------
extern "C" void kernel_launch(void* const* d_in, const int* in_sizes, int n_in,
                              void* d_out, int out_size, void* d_ws, size_t ws_size,
                              hipStream_t stream) {
  const float* xc_off  = (const float*)d_in[0];  // (B,U,2)
  const float* xc_on   = (const float*)d_in[1];  // (B,N0,N1,2)
  const float* z_off   = (const float*)d_in[2];  // (B,U,E)
  const float* latents = (const float*)d_in[4];  // (S,E)
  const float* fake    = (const float*)d_in[5];  // (E,)
  const float* Wq = (const float*)d_in[6];
  const float* Wk = (const float*)d_in[7];
  const float* Wv = (const float*)d_in[8];
  const float* Wo = (const float*)d_in[9];
  float* out = (float*)d_out;  // [xc_on_grid (B*S*2) | z_grid (B*S*E)]

  float* ws     = (float*)d_ws;
  float* QH     = ws;                               // S*E
  float* Kp     = QH + (size_t)S_ * E_;             // NTOK*E
  float* Vp     = Kp + (size_t)NTOK * E_;           // NTOK*E
  float* logits = Vp + (size_t)NTOK * E_;           // NTOK*H
  float* flog   = logits + (size_t)NTOK * H_;       // S*H
  float* kfake  = flog + (size_t)S_ * H_;           // E
  float* vfake  = kfake + E_;                       // E
  float* OutH   = vfake + E_;                       // NCELL*E
  int* counts   = (int*)(OutH + (size_t)NCELL * E_);// NCELL
  int* ctok     = counts + NCELL;                   // NCELL*MAXP
  int* cellid   = ctok + (size_t)NCELL * MAXP;      // NTOK

  // counts must be zero each call (ws is not re-poisoned between replays)
  k_zero<<<NCELL / 256, 256, 0, stream>>>(counts, NCELL);
  // pass-through first output (vectorized)
  k_copy4<<<(B_ * S_ * 2 / 4) / 256, 256, 0, stream>>>(
      (const float4*)xc_on, (float4*)out, B_ * S_ * 2 / 4);
  // binning
  k_bin<<<NTOK / 256, 256, 0, stream>>>(xc_off, counts, ctok, cellid);
  // fake token projections
  k_fake<<<1, 128, 0, stream>>>(fake, Wk, Wv, kfake, vfake);
  // WMMA GEMMs: QH = latents@Wq (batch-shared); K,V fused dual-W pass
  gemm_f16_128<<<64, 128, 0, stream>>>(latents, Wq, QH, S_ / 64);
  gemm_f16_128_dual<<<256, 128, 0, stream>>>(z_off, Wk, Wv, Kp, Vp, NTOK / 64);
  // attention logits
  k_flogit<<<(S_ * H_) / 256, 256, 0, stream>>>(QH, kfake, flog);
  k_logits<<<(NTOK * H_) / 256, 256, 0, stream>>>(QH, Kp, cellid, logits);
  // per-cell softmax + weighted V
  k_attn<<<NCELL / 8, 256, 0, stream>>>(counts, ctok, logits, flog, Vp, vfake, OutH);
  // output projection straight into d_out
  gemm_f16_128<<<256, 128, 0, stream>>>(OutH, Wo, out + B_ * S_ * 2, NCELL / 64);
}
